// Attention_74191265071167
// MI455X (gfx1250) — matrix operand order repaired and checked by reading, not compiled, not run on hardware
//
#include <hip/hip_runtime.h>
#include <hip/hip_bf16.h>
#include <cmath>
#include <cstdint>

#define B_      32
#define N_      576
#define DIM_    768
#define HEADS_  12
#define DH_     64
#define INNER_  768
#define QKV_LD  2304        // [Q | K | V] columns
#define SCALE_F 0.125f      // 64^-0.5

typedef __attribute__((ext_vector_type(16))) __bf16 v16bf;
typedef __attribute__((ext_vector_type(8)))  float  v8f;
typedef __attribute__((ext_vector_type(4)))  float  v4f;

__device__ inline v8f wmma_bf16(v16bf a, v16bf b, v8f c) {
  return __builtin_amdgcn_wmma_f32_16x16x32_bf16(false, a, false, b, (short)0, c,
                                                 false, false);
}

// ---- fragment loaders (contiguous k = 16*h + e order, wave32) -----------------------
__device__ inline v16bf load_a_bf16(const __bf16* __restrict__ Ap, int lda) {
  const int lane = threadIdx.x & 31;
  const int m = lane & 15, h2 = lane >> 4;
  const __bf16* p = Ap + (size_t)m * lda;
  v16bf r;
#pragma unroll
  for (int e = 0; e < 16; ++e) r[e] = p[h2 * 16 + e];
  return r;
}
// B[k][n] = Bp[n*ldb + k]  (K-contiguous per column, 32B/lane)
__device__ inline v16bf load_b_cm_bf16(const __bf16* __restrict__ Bp, int ldb) {
  const int lane = threadIdx.x & 31;
  const int n = lane & 15, h2 = lane >> 4;
  const __bf16* p = Bp + (size_t)n * ldb;
  v16bf r;
#pragma unroll
  for (int e = 0; e < 16; ++e) r[e] = p[h2 * 16 + e];
  return r;
}
// row-major source: B[k][n] = Bp[k*ldb + n]  (for V tiles)
__device__ inline v16bf load_b_rm_bf16(const __bf16* __restrict__ Bp, int ldb) {
  const int lane = threadIdx.x & 31;
  const int n = lane & 15, h2 = lane >> 4;
  v16bf r;
#pragma unroll
  for (int e = 0; e < 16; ++e) r[e] = Bp[(size_t)(h2 * 16 + e) * ldb + n];
  return r;
}

// ---- async global->LDS staging (ASYNCcnt path) ----------------------------
// Flat shared-aperture address truncates to addr[31:0] == LDS byte offset
// (ISA §10.2), so the low 32 bits of a __shared__ pointer are the VDST value.
__device__ inline void async_stage_row64B(__bf16* ldsrow, const __bf16* grow) {
  const unsigned lds = (unsigned)(uintptr_t)ldsrow;
  const unsigned long long g = (unsigned long long)(uintptr_t)grow;
#pragma unroll
  for (int c = 0; c < 4; ++c)
    asm volatile("global_load_async_to_lds_b128 %0, %1, off"
                 :: "v"(lds + c * 16), "v"(g + (unsigned long long)(c * 16))
                 : "memory");
}

// ---- shared 128x128-block GEMM mainloop (4 waves, LDS double-buffered) ----
// A: [M][K] row-major bf16; Bt: [Ncols][K] (K-contiguous) bf16; K = 768.
__device__ inline void gemm128_mainloop(const __bf16* __restrict__ Ag,
                                        const __bf16* __restrict__ Bg,
                                        int m0, int n0, int K,
                                        __bf16 (*sA)[128][40],
                                        __bf16 (*sB)[128][40],
                                        v8f (&acc)[4][4]) {
  const int tid = threadIdx.y * 32 + threadIdx.x;          // 0..127
  const int wr = (threadIdx.y >> 1) * 64, wc = (threadIdx.y & 1) * 64;
  // prologue: stage k-slice 0 into buffer 0
  async_stage_row64B(&sA[0][tid][0], Ag + (size_t)(m0 + tid) * K);
  async_stage_row64B(&sB[0][tid][0], Bg + (size_t)(n0 + tid) * K);
  int buf = 0;
  for (int k0 = 0; k0 < K; k0 += 32) {
    if (k0 + 32 < K) {
      async_stage_row64B(&sA[buf ^ 1][tid][0],
                         Ag + (size_t)(m0 + tid) * K + k0 + 32);
      async_stage_row64B(&sB[buf ^ 1][tid][0],
                         Bg + (size_t)(n0 + tid) * K + k0 + 32);
      // 8 ops just issued for next slice; <=8 outstanding => current resident
      asm volatile("s_wait_asynccnt 8" ::: "memory");
    } else {
      asm volatile("s_wait_asynccnt 0" ::: "memory");
    }
    __syncthreads();
    v16bf a[4], bb[4];
#pragma unroll
    for (int i = 0; i < 4; ++i) a[i] = load_a_bf16(&sA[buf][wr + 16 * i][0], 40);
#pragma unroll
    for (int j = 0; j < 4; ++j)
      bb[j] = load_b_cm_bf16(&sB[buf][wc + 16 * j][0], 40);
#pragma unroll
    for (int i = 0; i < 4; ++i)
#pragma unroll
      for (int j = 0; j < 4; ++j) acc[i][j] = wmma_bf16(a[i], bb[j], acc[i][j]);
    __syncthreads();   // all waves done reading buf before it is restaged
    buf ^= 1;
  }
}

// ---- prep kernel A: bf16-transpose the weights ----------------------------
__global__ __launch_bounds__(256) void prep_w_kernel(
    const float* __restrict__ Wq, const float* __restrict__ Wkv,
    const float* __restrict__ Wout, void* __restrict__ wqkvT_raw,
    void* __restrict__ woutT_raw) {
  __bf16* wqkvT = (__bf16*)wqkvT_raw;
  __bf16* woutT = (__bf16*)woutT_raw;
  const size_t tid = (size_t)blockIdx.x * 256 + threadIdx.x;
  if (tid < (size_t)QKV_LD * DIM_) {
    const int c = (int)(tid / DIM_), k = (int)(tid % DIM_);
    const float v = (c < INNER_) ? Wq[(size_t)k * INNER_ + c]
                                 : Wkv[(size_t)k * (2 * INNER_) + (c - INNER_)];
    wqkvT[tid] = (__bf16)v;
  }
  if (tid < (size_t)DIM_ * INNER_) {
    const int c = (int)(tid / INNER_), k = (int)(tid % INNER_);
    woutT[tid] = (__bf16)Wout[(size_t)k * DIM_ + c];
  }
}

// ---- prep kernel B: x -> bf16 ---------------------------------------------
__global__ __launch_bounds__(256) void prep_x_kernel(
    const float* __restrict__ x, void* __restrict__ xbf_raw) {
  __bf16* xbf = (__bf16*)xbf_raw;
  const size_t i0 = ((size_t)blockIdx.x * 256 + threadIdx.x) * 4;
#pragma unroll
  for (int e = 0; e < 4; ++e) xbf[i0 + e] = (__bf16)x[i0 + e];
}

// ---- kernel 1: QKV projection (128x128 block, async-LDS staged) -----------
__global__ __launch_bounds__(128) void qkv_proj_kernel(
    const void* __restrict__ xbf_raw, const void* __restrict__ wqkvT_raw,
    void* __restrict__ qkv_raw) {
  const __bf16* xbf = (const __bf16*)xbf_raw;
  const __bf16* wqkvT = (const __bf16*)wqkvT_raw;
  __bf16* qkv = (__bf16*)qkv_raw;
  __shared__ __bf16 sA[2][128][40];
  __shared__ __bf16 sB[2][128][40];
  const int tiles_n = QKV_LD / 128;                // 18
  const int m0 = (blockIdx.x / tiles_n) * 128;
  const int n0 = (blockIdx.x % tiles_n) * 128;
  v8f acc[4][4] = {};
  gemm128_mainloop(xbf, wqkvT, m0, n0, DIM_, sA, sB, acc);
  const int lane = threadIdx.x & 31;
  const int n = lane & 15, h2 = lane >> 4;
  const int wr = (threadIdx.y >> 1) * 64, wc = (threadIdx.y & 1) * 64;
#pragma unroll
  for (int i = 0; i < 4; ++i)
#pragma unroll
    for (int j = 0; j < 4; ++j) {
      const int col = n0 + wc + 16 * j + n;
      const float sc = (col < INNER_) ? SCALE_F : 1.0f;   // fold scale into Q
#pragma unroll
      for (int r = 0; r < 8; ++r) {
        const int row = m0 + wr + 16 * i + r + 8 * h2;
        qkv[(size_t)row * QKV_LD + col] = (__bf16)(acc[i][j][r] * sc);
      }
    }
}

// ---- kernel 2: talking-heads pre-mix + online softmax stats ---------------
__global__ __launch_bounds__(384) void attn_stats_kernel(
    const void* __restrict__ qkv_raw, const float* __restrict__ mix_pre,
    float* __restrict__ ml) {
  const __bf16* qkv = (const __bf16*)qkv_raw;
  __shared__ float sT[16][32 * HEADS_];   // [m][j][h] head-innermost
  const int b = blockIdx.y, i0 = blockIdx.x * 16;
  const int g = threadIdx.y;
  const int lane = threadIdx.x;
  const int n = lane & 15, h2 = lane >> 4;
  float wpre[HEADS_];
#pragma unroll
  for (int h = 0; h < HEADS_; ++h) wpre[h] = mix_pre[h * HEADS_ + g];

  v16bf qa[2];
  const __bf16* qbase = qkv + (size_t)(b * N_ + i0) * QKV_LD + g * DH_;
  qa[0] = load_a_bf16(qbase, QKV_LD);
  qa[1] = load_a_bf16(qbase + 32, QKV_LD);
  float mrun[8], lrun[8];
#pragma unroll
  for (int r = 0; r < 8; ++r) { mrun[r] = -INFINITY; lrun[r] = 0.f; }

  for (int j0 = 0; j0 < N_; j0 += 32) {
    v8f s[2];
#pragma unroll
    for (int jt = 0; jt < 2; ++jt) {
      v8f a = {};
      const __bf16* kb =
          qkv + (size_t)(b * N_ + j0 + jt * 16) * QKV_LD + INNER_ + g * DH_;
      a = wmma_bf16(qa[0], load_b_cm_bf16(kb, QKV_LD), a);
      a = wmma_bf16(qa[1], load_b_cm_bf16(kb + 32, QKV_LD), a);
      s[jt] = a;
    }
    if (j0 + 32 < N_)  // prefetch next K tile: one 128B line per lane
      __builtin_prefetch(qkv + (size_t)(b * N_ + j0 + 32 + lane) * QKV_LD +
                             INNER_ + g * DH_, 0, 0);
    __syncthreads();
#pragma unroll
    for (int r = 0; r < 8; ++r) {
      const int m = r + 8 * h2;
      sT[m][n * HEADS_ + g] = s[0][r];
      sT[m][(16 + n) * HEADS_ + g] = s[1][r];
    }
    __syncthreads();
    float s0[8], s1[8];
#pragma unroll
    for (int r = 0; r < 8; ++r) {
      const int m = r + 8 * h2;
#pragma unroll
      for (int half = 0; half < 2; ++half) {
        const float* pp = &sT[m][(half * 16 + n) * HEADS_];
        const v4f q0 = *(const v4f*)pp;
        const v4f q1 = *(const v4f*)(pp + 4);
        const v4f q2 = *(const v4f*)(pp + 8);
        float a = 0.f;
#pragma unroll
        for (int h = 0; h < 4; ++h) {
          a += wpre[h] * q0[h];
          a += wpre[4 + h] * q1[h];
          a += wpre[8 + h] * q2[h];
        }
        if (half == 0) s0[r] = a; else s1[r] = a;
      }
    }
#pragma unroll
    for (int r = 0; r < 8; ++r) {
      float tmax = fmaxf(s0[r], s1[r]);
#pragma unroll
      for (int msk = 8; msk >= 1; msk >>= 1)
        tmax = fmaxf(tmax, __shfl_xor(tmax, msk, 32));
      const float mnew = fmaxf(mrun[r], tmax);
      float psum = __expf(s0[r] - mnew) + __expf(s1[r] - mnew);
#pragma unroll
      for (int msk = 8; msk >= 1; msk >>= 1) psum += __shfl_xor(psum, msk, 32);
      lrun[r] = lrun[r] * __expf(mrun[r] - mnew) + psum;
      mrun[r] = mnew;
    }
  }
  if (n == 0) {
#pragma unroll
    for (int r = 0; r < 8; ++r) {
      const int row = i0 + r + 8 * h2;
      const size_t idx = (((size_t)b * HEADS_ + g) * N_ + row) * 2;
      ml[idx] = mrun[r];
      ml[idx + 1] = lrun[r];
    }
  }
}

// ---- kernel 3: recompute dots, exact softmax, post-mix, PV ----------------
#define PROW 528   // 32*16 + 16 pad: m-stride = 264 dwords == 8 mod 64
__global__ __launch_bounds__(384) void attn_out_kernel(
    const void* __restrict__ qkv_raw, const float* __restrict__ mix_pre,
    const float* __restrict__ mix_post, const float* __restrict__ ml,
    void* __restrict__ oinner_raw) {
  const __bf16* qkv = (const __bf16*)qkv_raw;
  __bf16* oinner = (__bf16*)oinner_raw;
  __shared__ float  sT[16][32 * HEADS_];   // raw dots, [m][j][h]
  __shared__ __bf16 sPT[16][PROW];         // normalized attn, [m][j*16+h]
  const int b = blockIdx.y, i0 = blockIdx.x * 16;
  const int g = threadIdx.y;               // wave = head (both roles)
  const int lane = threadIdx.x;
  const int n = lane & 15, h2 = lane >> 4;
  float wpre[HEADS_], wpost[HEADS_];
#pragma unroll
  for (int h = 0; h < HEADS_; ++h) {
    wpre[h] = mix_pre[h * HEADS_ + g];
    wpost[h] = mix_post[h * HEADS_ + g];
  }
  v16bf qa[2];
  const __bf16* qbase = qkv + (size_t)(b * N_ + i0) * QKV_LD + g * DH_;
  qa[0] = load_a_bf16(qbase, QKV_LD);
  qa[1] = load_a_bf16(qbase + 32, QKV_LD);
  float mfin[8], invl[8];
#pragma unroll
  for (int r = 0; r < 8; ++r) {
    const int row = i0 + r + 8 * h2;
    const size_t idx = (((size_t)b * HEADS_ + g) * N_ + row) * 2;
    mfin[r] = ml[idx];
    invl[r] = 1.0f / ml[idx + 1];
  }
  v8f oacc[4] = {};

  for (int j0 = 0; j0 < N_; j0 += 32) {
    v8f s[2];
#pragma unroll
    for (int jt = 0; jt < 2; ++jt) {
      v8f a = {};
      const __bf16* kb =
          qkv + (size_t)(b * N_ + j0 + jt * 16) * QKV_LD + INNER_ + g * DH_;
      a = wmma_bf16(qa[0], load_b_cm_bf16(kb, QKV_LD), a);
      a = wmma_bf16(qa[1], load_b_cm_bf16(kb + 32, QKV_LD), a);
      s[jt] = a;
    }
    if (j0 + 32 < N_) {  // prefetch next K and V tiles: one row per lane
      const __bf16* nrow = qkv + (size_t)(b * N_ + j0 + 32 + lane) * QKV_LD;
      __builtin_prefetch(nrow + INNER_ + g * DH_, 0, 0);
      __builtin_prefetch(nrow + 2 * INNER_ + g * DH_, 0, 0);
    }
    __syncthreads();
#pragma unroll
    for (int r = 0; r < 8; ++r) {
      const int m = r + 8 * h2;
      sT[m][n * HEADS_ + g] = s[0][r];
      sT[m][(16 + n) * HEADS_ + g] = s[1][r];
    }
    __syncthreads();
#pragma unroll
    for (int r = 0; r < 8; ++r) {
      const int m = r + 8 * h2;
#pragma unroll
      for (int half = 0; half < 2; ++half) {
        const float* pp = &sT[m][(half * 16 + n) * HEADS_];
        const v4f q0 = *(const v4f*)pp;
        const v4f q1 = *(const v4f*)(pp + 4);
        const v4f q2 = *(const v4f*)(pp + 8);
        float a = 0.f;
#pragma unroll
        for (int h = 0; h < 4; ++h) {
          a += wpre[h] * q0[h];
          a += wpre[4 + h] * q1[h];
          a += wpre[8 + h] * q2[h];
        }
        sPT[m][(half * 16 + n) * 16 + g] =
            (__bf16)(__expf(a - mfin[r]) * invl[r]);
      }
    }
    __syncthreads();
    // wave g now acts as OUTPUT head e=g: post-mix + rebuild A-fragment
    v16bf pfrag;
    const int mA = lane & 15;
#pragma unroll
    for (int e2 = 0; e2 < 16; ++e2) {
      const int jl = h2 * 16 + e2;
      const v16bf pv = *(const v16bf*)&sPT[mA][jl * 16];  // 2x ds_load_b128
      float v = 0.f;
#pragma unroll
      for (int h = 0; h < HEADS_; ++h) v += wpost[h] * (float)pv[h];
      pfrag[e2] = (__bf16)v;
    }
#pragma unroll
    for (int dc = 0; dc < 4; ++dc) {
      const __bf16* vb = qkv + (size_t)(b * N_ + j0) * QKV_LD + 2 * INNER_ +
                         g * DH_ + dc * 16;
      oacc[dc] = wmma_bf16(pfrag, load_b_rm_bf16(vb, QKV_LD), oacc[dc]);
    }
  }
#pragma unroll
  for (int dc = 0; dc < 4; ++dc)
#pragma unroll
    for (int r = 0; r < 8; ++r) {
      const int row = b * N_ + i0 + r + 8 * h2;
      const int col = g * DH_ + dc * 16 + n;
      oinner[(size_t)row * INNER_ + col] = (__bf16)oacc[dc][r];
    }
}

// ---- kernel 4: output projection (128x128 block, async-LDS staged) --------
__global__ __launch_bounds__(128) void out_proj_kernel(
    const void* __restrict__ oinner_raw, const void* __restrict__ woutT_raw,
    const float* __restrict__ bout, float* __restrict__ out) {
  const __bf16* oin = (const __bf16*)oinner_raw;
  const __bf16* woutT = (const __bf16*)woutT_raw;
  __shared__ __bf16 sA[2][128][40];
  __shared__ __bf16 sB[2][128][40];
  const int tiles_n = DIM_ / 128;                  // 6
  const int m0 = (blockIdx.x / tiles_n) * 128;
  const int n0 = (blockIdx.x % tiles_n) * 128;
  v8f acc[4][4] = {};
  gemm128_mainloop(oin, woutT, m0, n0, INNER_, sA, sB, acc);
  const int lane = threadIdx.x & 31;
  const int n = lane & 15, h2 = lane >> 4;
  const int wr = (threadIdx.y >> 1) * 64, wc = (threadIdx.y & 1) * 64;
#pragma unroll
  for (int i = 0; i < 4; ++i)
#pragma unroll
    for (int j = 0; j < 4; ++j) {
      const int col = n0 + wc + 16 * j + n;
      const float bias = bout[col];
#pragma unroll
      for (int r = 0; r < 8; ++r) {
        const int row = m0 + wr + 16 * i + r + 8 * h2;
        out[(size_t)row * DIM_ + col] = acc[i][j][r] + bias;
      }
    }
}

extern "C" void kernel_launch(void* const* d_in, const int* in_sizes, int n_in,
                              void* d_out, int out_size, void* d_ws,
                              size_t ws_size, hipStream_t stream) {
  (void)in_sizes; (void)n_in; (void)out_size; (void)ws_size;
  const float* x        = (const float*)d_in[0];
  const float* Wq       = (const float*)d_in[1];
  const float* Wkv      = (const float*)d_in[2];
  const float* mix_pre  = (const float*)d_in[3];
  const float* mix_post = (const float*)d_in[4];
  const float* Wout     = (const float*)d_in[5];
  const float* bout     = (const float*)d_in[6];
  float* out = (float*)d_out;

  char* ws = (char*)d_ws;
  const size_t rows = (size_t)B_ * N_;                       // 18432
  void*  qkv    = ws;                                        // bf16 [rows x 2304]
  size_t off    = rows * QKV_LD * 2;
  float* ml     = (float*)(ws + off);                        // f32 [B,H,N,2]
  off          += (size_t)B_ * HEADS_ * N_ * 2 * 4;
  void*  oinner = ws + off;                                  // bf16 [rows x 768]
  off          += rows * INNER_ * 2;
  void*  xbf    = ws + off;                                  // bf16 [rows x 768]
  off          += rows * DIM_ * 2;
  void*  wqkvT  = ws + off;                                  // bf16 [2304 x 768]
  off          += (size_t)QKV_LD * DIM_ * 2;
  void*  woutT  = ws + off;                                  // bf16 [768 x 768]

  dim3 blkg(32, 4);
  dim3 blka(32, HEADS_);
  dim3 grda(N_ / 16, B_);

  prep_w_kernel<<<(int)(((size_t)QKV_LD * DIM_ + 255) / 256), dim3(256), 0,
                  stream>>>(Wq, Wkv, Wout, wqkvT, woutT);
  prep_x_kernel<<<(int)((rows * DIM_ / 4 + 255) / 256), dim3(256), 0, stream>>>(
      x, xbf);
  qkv_proj_kernel<<<(int)((rows / 128) * (QKV_LD / 128)), blkg, 0, stream>>>(
      xbf, wqkvT, qkv);
  attn_stats_kernel<<<grda, blka, 0, stream>>>(qkv, mix_pre, ml);
  attn_out_kernel<<<grda, blka, 0, stream>>>(qkv, mix_pre, mix_post, ml, oinner);
  out_proj_kernel<<<(int)((rows / 128) * (DIM_ / 128)), blkg, 0, stream>>>(
      oinner, woutT, bout, out);
}
